// FANSNeuralStateUpdate_120259084384
// MI455X (gfx1250) — matrix operand order repaired and checked by reading, not compiled, not run on hardware
//
#include <hip/hip_runtime.h>
#include <hip/hip_bf16.h>

typedef __attribute__((ext_vector_type(16))) _Float16 v16h;
typedef __attribute__((ext_vector_type(8)))  float    v8f;

#define NWAVES 8
#define WGSIZE (NWAVES * 32)
#define MTILES 4   // batch tiles (of 16 rows) per wave, amortizes weight staging

// IDX[s][i] = sorted((s+i)%16, i=0..7) in closed form:
//   s <= 8 : s+i  (no wrap)
//   s >= 9 : first (s-8) entries are 0..s-9, rest are (i+8)
__device__ __forceinline__ int idx_sel(int s, int i) {
  return (s <= 8) ? (s + i) : ((i < s - 8) ? i : (i + 8));
}

// Branch-free tanh: CDNA5 hardware transcendental v_tanh_f32 (TRANS op,
// co-executes with XDL WMMA per ISA §8.4); exp2/rcp fallback keeps it
// single-path if the builtin is missing.
__device__ __forceinline__ float fast_tanh(float x) {
#if __has_builtin(__builtin_amdgcn_tanhf)
  return __builtin_amdgcn_tanhf(x);
#elif __has_builtin(__builtin_amdgcn_tanh_f32)
  return __builtin_amdgcn_tanh_f32(x);
#else
  const float ax = __builtin_fabsf(x);
  const float t  = __builtin_amdgcn_exp2f(ax * -2.8853900817779268f); // e^{-2|x|}
  const float r  = (1.0f - t) * __builtin_amdgcn_rcpf(1.0f + t);
  return __builtin_copysignf(r, x);
#endif
}

__global__ __launch_bounds__(WGSIZE)
void fans_neural_state_update_kernel(const float* __restrict__ x_f,
                                     const float* __restrict__ x_b,
                                     const float* __restrict__ u,
                                     const float* __restrict__ W0,
                                     const float* __restrict__ W1,
                                     const float* __restrict__ W2,
                                     float* __restrict__ out) {
  __shared__ _Float16 sW0[64 * 12];           // W0[s][o][i] -> sW0[o*12+i]
  __shared__ _Float16 sW1[64 * 64];           // W1[s][o][k] -> sW1[o*64+k]
  __shared__ _Float16 sW2[64];                // W2[s][0][k]
  __shared__ _Float16 hb0[NWAVES][16 * 64];   // per-wave h0 staging (row-major 16x64)
  __shared__ _Float16 hb1[NWAVES][16 * 64];   // per-wave h1 staging

  const int tid  = threadIdx.x;
  const int wave = tid >> 5;
  const int lane = tid & 31;
  const int s    = blockIdx.y;                               // state index 0..15
  const int m0   = (blockIdx.x * NWAVES + wave) * (16 * MTILES);

  // ---- cooperative weight staging: f32 global -> f16 LDS (once per block) ----
  const float* W0s = W0 + s * 64 * 12;
  const float* W1s = W1 + s * 64 * 64;
  const float* W2s = W2 + s * 64;
  for (int i = tid; i < 64 * 12; i += WGSIZE) sW0[i] = (_Float16)W0s[i];
  for (int i = tid; i < 64 * 64; i += WGSIZE) sW1[i] = (_Float16)W1s[i];
  if (tid < 64) sW2[tid] = (_Float16)W2s[tid];
  __syncthreads();

  const int row = lane & 15;   // M row (and N col for B/C/D fragments)
  const int hi  = lane >> 4;   // K-half selector per ISA A/B layouts
  const int kb  = hi * 8;

  _Float16* hw0 = hb0[wave];
  _Float16* hw1 = hb1[wave];

  // ---- hoist tile-invariant B fragments into registers ----
  // Layer-1 W1: per N-tile, two K-step fragments (K 0..31 and 32..63).
  // B layout 32x16 f16: lanes 0-15 hold K=0..7 & 16..23, lanes 16-31 hold K=8..15 & 24..31.
  v16h w1A[4], w1B[4];
#pragma unroll
  for (int nt = 0; nt < 4; ++nt) {
    const int n = nt * 16 + row;
#pragma unroll
    for (int e = 0; e < 16; ++e) {
      w1A[nt][e] = sW1[n * 64 +      hi * 8 + (e & 7) + (e >> 3) * 16];
      w1B[nt][e] = sW1[n * 64 + 32 + hi * 8 + (e & 7) + (e >> 3) * 16];
    }
  }
  // Layer-2 W2: only column N=0 populated (lanes 0 and 16).
  v16h w20{}, w21{};
  if (row == 0) {
#pragma unroll
    for (int e = 0; e < 16; ++e) {
      w20[e] = sW2[     hi * 8 + (e & 7) + (e >> 3) * 16];
      w21[e] = sW2[32 + hi * 8 + (e & 7) + (e >> 3) * 16];
    }
  }

  for (int mt = 0; mt < MTILES; ++mt) {
    const int b = m0 + mt * 16 + row;   // global batch row for this lane

    // ---- layer 0 A fragment: z = [x_sel(8) | u(4) | pad..K=32] ----
    // A layout 16x32 f16: lane(0-15) K=0..7 & 16..23 ; lane(16-31) K=8..15 & 24..31.
    // zin=12 => lanes 0-15 carry x_sel (K 0..7); lanes 16-31 carry u (K 8..11).
    v16h a0{};
    if (hi == 0) {
#pragma unroll
      for (int i = 0; i < 8; ++i) {
        const int j = idx_sel(s, i);
        const float v = (j < 8) ? x_f[b * 8 + j] : x_b[b * 8 + (j - 8)];
        a0[i] = (_Float16)v;
      }
    } else {
#pragma unroll
      for (int i = 0; i < 4; ++i) a0[i] = (_Float16)u[b * 4 + i];
    }

    // ---- layer 0: h0 = tanh(z @ W0^T), N=64 as 4 tiles of 16 ----
#pragma unroll
    for (int nt = 0; nt < 4; ++nt) {
      const int n = nt * 16 + row;
      v16h bfrag{};
      {
#pragma unroll
        for (int e = 0; e < 8 - 4 * hi; ++e) bfrag[e] = sW0[n * 12 + hi * 8 + e];
      }
      v8f c{};
      c = __builtin_amdgcn_wmma_f32_16x16x32_f16(false, a0, false, bfrag,
                                                 (short)0, c, false, false);
      // D layout: VGPR r -> M = r + 8*hi, N = nt*16 + row
#pragma unroll
      for (int r = 0; r < 8; ++r)
        hw0[(r + hi * 8) * 64 + nt * 16 + row] = (_Float16)fast_tanh(c[r]);
    }

    // ---- layer 1 A fragments (K=64 -> two 16x32 fragments) from hw0 ----
    v16h a10{}, a11{};
#pragma unroll
    for (int t = 0; t < 8; ++t) {
      a10[t]     = hw0[row * 64 +      kb + t];
      a10[8 + t] = hw0[row * 64 + 16 + kb + t];
      a11[t]     = hw0[row * 64 + 32 + kb + t];
      a11[8 + t] = hw0[row * 64 + 48 + kb + t];
    }

    // ---- layer 1: h1 = tanh(h0 @ W1^T) ----
#pragma unroll
    for (int nt = 0; nt < 4; ++nt) {
      v8f c{};
      c = __builtin_amdgcn_wmma_f32_16x16x32_f16(false, a10, false, w1A[nt],
                                                 (short)0, c, false, false);
      c = __builtin_amdgcn_wmma_f32_16x16x32_f16(false, a11, false, w1B[nt],
                                                 (short)0, c, false, false);
#pragma unroll
      for (int r = 0; r < 8; ++r)
        hw1[(r + hi * 8) * 64 + nt * 16 + row] = (_Float16)fast_tanh(c[r]);
    }

    // ---- layer 2: dx = h1 @ W2^T (64 -> 1), WMMA with only N=0 populated ----
    v16h a20{}, a21{};
#pragma unroll
    for (int t = 0; t < 8; ++t) {
      a20[t]     = hw1[row * 64 +      kb + t];
      a20[8 + t] = hw1[row * 64 + 16 + kb + t];
      a21[t]     = hw1[row * 64 + 32 + kb + t];
      a21[8 + t] = hw1[row * 64 + 48 + kb + t];
    }
    v8f c2{};
    c2 = __builtin_amdgcn_wmma_f32_16x16x32_f16(false, a20, false, w20,
                                                (short)0, c2, false, false);
    c2 = __builtin_amdgcn_wmma_f32_16x16x32_f16(false, a21, false, w21,
                                                (short)0, c2, false, false);

    // Column N=0 of D lives in lane 0 (M=0..7) and lane 16 (M=8..15)
    if (row == 0) {
      const int mb = m0 + mt * 16 + hi * 8;
#pragma unroll
      for (int r = 0; r < 8; ++r)
        out[(mb + r) * 16 + s] = c2[r];
    }
  }
}

extern "C" void kernel_launch(void* const* d_in, const int* in_sizes, int n_in,
                              void* d_out, int out_size, void* d_ws, size_t ws_size,
                              hipStream_t stream) {
  const float* x_f = (const float*)d_in[0];
  const float* x_b = (const float*)d_in[1];
  const float* u   = (const float*)d_in[2];
  const float* W0  = (const float*)d_in[3];
  const float* W1  = (const float*)d_in[4];
  const float* W2  = (const float*)d_in[5];
  float* out = (float*)d_out;

  const int B = in_sizes[0] / 8;                        // x_f is (B, 8)
  dim3 grid(B / (16 * NWAVES * MTILES), 16, 1);         // (batch super-tiles, states)
  fans_neural_state_update_kernel<<<grid, WGSIZE, 0, stream>>>(
      x_f, x_b, u, W0, W1, W2, out);
}
